// CrossPSDLoss_5781025980854
// MI455X (gfx1250) — matrix operand order repaired and checked by reading, compiled, not run on hardware
//
#include <hip/hip_runtime.h>
#include <hip/hip_bf16.h>

// ---------------------------------------------------------------------------
// CrossPSDLoss on MI455X (gfx1250, wave32, WMMA + async-LDS DMA).
//
// Masked-bin Welch PSD == GEMM against a DFT basis:
//   frames:  M = 2(psd) x 8(rows 8..15) x 512(segs, 511 real + 1 zero pad)
//   basis:   K = 4096 taps, N = 960 cols (479 freqs x {cos,sin}, 2 pads)
//   psd[row,f] = sum_j (Re^2+Im^2)  ==> accumulate sum of C^2 per column,
//   pair cos/sin at the end; 1/T cancels in the ratio.
// Precision: split f32 -> hi/lo bf16, 3 WMMAs (hh+hl+lh) ~ f32 accuracy on
// the high-throughput v_wmma_f32_16x16x32_bf16 path.
// Fast path: precompute split frames into ws (128 MB, L2-resident), then
// stage tiles with GLOBAL_LOAD_ASYNC_TO_LDS_B128 (ASYNCcnt) into
// double-buffered LDS so DMA overlaps the WMMA stream.
// ---------------------------------------------------------------------------

typedef __attribute__((ext_vector_type(16))) __bf16 v16bf;
typedef __attribute__((ext_vector_type(8)))  float  v8f;

union Frag16 { v16bf v; uint4 q[2]; };
union Pack16 { __bf16 h[16]; uint4 q[2]; };

#define WS_OFF_BHI   0ull
#define WS_OFF_BLO   7864320ull           // 960*4096*2
#define WS_OFF_S     15728640ull          // + 960*4096*2
#define WS_OFF_WIN   15790080ull          // + 2*8*960*4  (S = 15360 floats)
#define WS_OFF_AHI   15806464ull          // + 4096*4     (aligned to 256)
#define WS_OFF_ALO   82915328ull          // + 8192*4096*2
#define WS_NEED_PRE  150024192ull         // + 8192*4096*2  (~143.1 MiB)
// fallback (fused) path needs only 15,806,464 bytes (~15.1 MiB)

// Async DMA: global -> LDS, 16B per lane, tracked by ASYNCcnt (no VGPR data).
// lds  = LDS byte offset (low 32 bits of the generic address)
// voff = per-lane 32-bit byte offset, base = uniform 64-bit pointer (SGPR pair)
__device__ __forceinline__ void async_b128(unsigned lds, unsigned voff,
                                           const void* base) {
    asm volatile("global_load_async_to_lds_b128 %0, %1, %2"
                 :: "v"(lds), "v"(voff), "s"(base) : "memory");
}
__device__ __forceinline__ void wait_async0() {
    asm volatile("s_wait_asynccnt 0x0" ::: "memory");
}

// --- gather one 16-sample chunk of the windowed frame matrix ---------------
// m in [0,8192): psd(1b) | row(3b) | seg(9b);  k0 = 16-aligned tap index.
__device__ __forceinline__ void gather_frame16(const float* __restrict__ pred,
                                               const float* __restrict__ target,
                                               const float* __restrict__ win,
                                               int m, int k0, float a[16]) {
    const int psd  = m >> 12;            // 0 = res (target-pred), 1 = target
    const int i16  = ((m >> 9) & 7) + 8; // cross row 8..15
    const int j    = m & 511;            // segment (511 == zero pad)
    if (j == 511) {
#pragma unroll
        for (int q = 0; q < 16; ++q) a[q] = 0.0f;
        return;
    }
    const int pos = j * 2048 + k0;       // sample index in cross row
    const int b   = pos >> 10;           // batch
    const int t   = pos & 1023;          // offset within 1024-chunk (16-aligned)
    const size_t idx = (size_t)b * 16384 + (size_t)i16 * 1024 + t;
    const float4* tg4 = reinterpret_cast<const float4*>(target + idx);
    float4 t0 = tg4[0], t1 = tg4[1], t2 = tg4[2], t3 = tg4[3];
    a[0]=t0.x; a[1]=t0.y; a[2]=t0.z; a[3]=t0.w;
    a[4]=t1.x; a[5]=t1.y; a[6]=t1.z; a[7]=t1.w;
    a[8]=t2.x; a[9]=t2.y; a[10]=t2.z; a[11]=t2.w;
    a[12]=t3.x; a[13]=t3.y; a[14]=t3.z; a[15]=t3.w;
    if (psd == 0) {
        const float4* pr4 = reinterpret_cast<const float4*>(pred + idx);
        float4 p0 = pr4[0], p1 = pr4[1], p2 = pr4[2], p3 = pr4[3];
        a[0]-=p0.x; a[1]-=p0.y; a[2]-=p0.z; a[3]-=p0.w;
        a[4]-=p1.x; a[5]-=p1.y; a[6]-=p1.z; a[7]-=p1.w;
        a[8]-=p2.x; a[9]-=p2.y; a[10]-=p2.z; a[11]-=p2.w;
        a[12]-=p3.x; a[13]-=p3.y; a[14]-=p3.z; a[15]-=p3.w;
    }
    const float4* w4 = reinterpret_cast<const float4*>(win + k0);
    float4 w0 = w4[0], w1 = w4[1], w2 = w4[2], w3 = w4[3];
    a[0]*=w0.x; a[1]*=w0.y; a[2]*=w0.z; a[3]*=w0.w;
    a[4]*=w1.x; a[5]*=w1.y; a[6]*=w1.z; a[7]*=w1.w;
    a[8]*=w2.x; a[9]*=w2.y; a[10]*=w2.z; a[11]*=w2.w;
    a[12]*=w3.x; a[13]*=w3.y; a[14]*=w3.z; a[15]*=w3.w;
}

__device__ __forceinline__ void split16(const float a[16], Pack16& Ph, Pack16& Pl) {
#pragma unroll
    for (int q = 0; q < 16; ++q) {
        float v = a[q];
        __bf16 hv = (__bf16)v;
        Ph.h[q] = hv;
        Pl.h[q] = (__bf16)(v - (float)hv);
    }
}

// --- kernel 1: zero PSD accumulators, build window table -------------------
__global__ void init_ws_kernel(float* __restrict__ S, float* __restrict__ win) {
    int t = blockIdx.x * 256 + threadIdx.x;
    if (t < 15360) S[t] = 0.0f;
    if (t < 4096)  win[t] = 1.0f - cospif((float)t * (1.0f / 2048.0f));
}

// --- kernel 2: DFT basis, transposed [col][k], split hi/lo bf16 ------------
__global__ void build_basis_kernel(__bf16* __restrict__ bhi,
                                   __bf16* __restrict__ blo) {
    unsigned gid = blockIdx.x * 256u + threadIdx.x;   // 960*4096 threads exact
    int c = (int)(gid >> 12);          // 0..959
    int k = (int)(gid & 4095u);
    float v = 0.0f;
    if (c < 958) {
        int f = 21 + (c >> 1);                          // freq bin 21..499
        float a = (float)(f * k) * (1.0f / 2048.0f);    // exact: f*k < 2^24
        v = (c & 1) ? sinpif(a) : cospif(a);            // sign irrelevant (^2)
    }
    __bf16 h = (__bf16)v;
    __bf16 l = (__bf16)(v - (float)h);
    bhi[gid] = h;
    blo[gid] = l;
}

// --- kernel 2b: precompute split windowed frames (one m-row per block) -----
__global__ __launch_bounds__(256)
void prep_frames_kernel(const float* __restrict__ pred,
                        const float* __restrict__ target,
                        const float* __restrict__ win,
                        __bf16* __restrict__ Ahi, __bf16* __restrict__ Alo) {
    const int m  = blockIdx.x;              // 0..8191
    const int k0 = threadIdx.x * 16;        // 0..4080
    float a[16];
    gather_frame16(pred, target, win, m, k0, a);
    Pack16 Ph, Pl;
    split16(a, Ph, Pl);
    const size_t off = (size_t)m * 4096 + k0;
    *(uint4*)(Ahi + off)     = Ph.q[0];
    *(uint4*)(Ahi + off + 8) = Ph.q[1];
    *(uint4*)(Alo + off)     = Pl.q[0];
    *(uint4*)(Alo + off + 8) = Pl.q[1];
}

// --- kernel 3: split-bf16 WMMA GEMM + squared accumulation -----------------
// grid = (15 N-tiles, 64 M-tiles), 256 threads (8 waves, 4x2 wave grid).
// Block tile: 128(M) x 64(N), K-step 32.  Wave tile: 32x32 (2x2 WMMA frags).
template <bool PRE>
__global__ __launch_bounds__(256)
void dft_gemm_kernel(const float* __restrict__ pred,
                     const float* __restrict__ target,
                     const __bf16* __restrict__ Bhi,
                     const __bf16* __restrict__ Blo,
                     const float* __restrict__ win,
                     const __bf16* __restrict__ Ahi,
                     const __bf16* __restrict__ Alo,
                     float* __restrict__ S) {
    constexpr int NB = PRE ? 2 : 1;               // LDS double-buffer for DMA
    __shared__ alignas(16) __bf16 lAh[NB][128][40];   // padded stride (banks)
    __shared__ alignas(16) __bf16 lAl[NB][128][40];
    __shared__ alignas(16) __bf16 lBh[NB][64][40];
    __shared__ alignas(16) __bf16 lBl[NB][64][40];

    const int tid  = threadIdx.x;
    const int lane = tid & 31;
    const int wave = tid >> 5;
    const int wm   = wave >> 1;             // 0..3
    const int wn   = wave & 1;              // 0..1
    const int la   = lane & 15;
    const int lh   = lane >> 4;

    const int nBase = blockIdx.x * 64;      // 0..896
    const int mBase = blockIdx.y * 128;     // 0..8064

    const int arow   = tid >> 1;            // 0..127
    const int achunk = (tid & 1) * 16;      // 0,16
    const int crow   = tid >> 2;            // 0..63
    const int cchunk = (tid & 3) * 8;       // 0,8,16,24

    v8f acc[2][2];
#pragma unroll
    for (int a0 = 0; a0 < 2; ++a0)
#pragma unroll
        for (int a1 = 0; a1 < 2; ++a1)
#pragma unroll
            for (int e = 0; e < 8; ++e) acc[a0][a1][e] = 0.0f;

    // Frag-load + 12 split-precision WMMAs from LDS buffer `cur`.
    // A (16x32 bf16): lane holds row M=la; lanes<16: K 0..15,
    //                 lanes>=16: K 16..31 (contiguous).
    // B (32x16 bf16): lane holds col N=la; lanes<16: K 0..15,
    //                 lanes>=16: K 16..31 (contiguous).
    auto frag_wmma = [&](int cur) {
        Frag16 ah[2], al[2], bh[2], bl[2];
#pragma unroll
        for (int sm = 0; sm < 2; ++sm) {
            const int rA = wm * 32 + sm * 16 + la;
            const int k0 = lh * 16;
            ah[sm].q[0] = *(const uint4*)&lAh[cur][rA][k0];
            ah[sm].q[1] = *(const uint4*)&lAh[cur][rA][k0 + 8];
            al[sm].q[0] = *(const uint4*)&lAl[cur][rA][k0];
            al[sm].q[1] = *(const uint4*)&lAl[cur][rA][k0 + 8];
        }
#pragma unroll
        for (int sn = 0; sn < 2; ++sn) {
            const int rB = wn * 32 + sn * 16 + la;
            const int k0 = lh * 16;
            bh[sn].q[0] = *(const uint4*)&lBh[cur][rB][k0];
            bh[sn].q[1] = *(const uint4*)&lBh[cur][rB][k0 + 8];
            bl[sn].q[0] = *(const uint4*)&lBl[cur][rB][k0];
            bl[sn].q[1] = *(const uint4*)&lBl[cur][rB][k0 + 8];
        }
#pragma unroll
        for (int sm = 0; sm < 2; ++sm)
#pragma unroll
            for (int sn = 0; sn < 2; ++sn) {
                acc[sm][sn] = __builtin_amdgcn_wmma_f32_16x16x32_bf16(
                    false, ah[sm].v, false, bh[sn].v, (short)0, acc[sm][sn],
                    false, false);
                acc[sm][sn] = __builtin_amdgcn_wmma_f32_16x16x32_bf16(
                    false, ah[sm].v, false, bl[sn].v, (short)0, acc[sm][sn],
                    false, false);
                acc[sm][sn] = __builtin_amdgcn_wmma_f32_16x16x32_bf16(
                    false, al[sm].v, false, bh[sn].v, (short)0, acc[sm][sn],
                    false, false);
            }
    };

    if constexpr (PRE) {
        // ---- async-DMA double-buffered pipeline --------------------------
        const unsigned aBase = (unsigned)(((mBase + arow) * 4096 + achunk) * 2);
        const unsigned bBase = (unsigned)(((nBase + crow) * 4096 + cchunk) * 2);
        unsigned dAh[2], dAl[2], dBh[2], dBl[2];
#pragma unroll
        for (int bb = 0; bb < 2; ++bb) {     // LDS byte offsets per buffer
            dAh[bb] = (unsigned)(size_t)&lAh[bb][arow][achunk];
            dAl[bb] = (unsigned)(size_t)&lAl[bb][arow][achunk];
            dBh[bb] = (unsigned)(size_t)&lBh[bb][crow][cchunk];
            dBl[bb] = (unsigned)(size_t)&lBl[bb][crow][cchunk];
        }
        auto issue = [&](int bb, int kt) {   // 6 x 16B DMA per thread
            const unsigned vA = aBase + (unsigned)kt * 64u;
            const unsigned vB = bBase + (unsigned)kt * 64u;
            async_b128(dAh[bb],      vA,      Ahi);
            async_b128(dAh[bb] + 16, vA + 16, Ahi);
            async_b128(dAl[bb],      vA,      Alo);
            async_b128(dAl[bb] + 16, vA + 16, Alo);
            async_b128(dBh[bb],      vB,      Bhi);
            async_b128(dBl[bb],      vB,      Blo);
        };
        issue(0, 0);                          // prologue
        for (int kt = 0; kt < 128; ++kt) {
            const int cur = kt & 1;
            wait_async0();                    // this wave's buf[cur] landed
            __syncthreads();                  // all waves' writes visible;
                                              // prev reads of buf[cur^1] done
            if (kt < 127) issue(cur ^ 1, kt + 1);   // DMA overlaps WMMAs
            frag_wmma(cur);
        }
    } else {
        // ---- fused gather fallback (small workspace) ---------------------
        for (int kt = 0; kt < 128; ++kt) {
            const int kBase = kt * 32;
            float a[16];
            gather_frame16(pred, target, win, mBase + arow, kBase + achunk, a);
            Pack16 Ph, Pl;
            split16(a, Ph, Pl);
            const size_t boff = (size_t)(nBase + crow) * 4096 + kBase + cchunk;
            const uint4 bh0 = *(const uint4*)(Bhi + boff);
            const uint4 bl0 = *(const uint4*)(Blo + boff);

            __syncthreads();   // previous iteration's frags fully consumed
            *(uint4*)&lAh[0][arow][achunk]     = Ph.q[0];
            *(uint4*)&lAh[0][arow][achunk + 8] = Ph.q[1];
            *(uint4*)&lAl[0][arow][achunk]     = Pl.q[0];
            *(uint4*)&lAl[0][arow][achunk + 8] = Pl.q[1];
            *(uint4*)&lBh[0][crow][cchunk]     = bh0;
            *(uint4*)&lBl[0][crow][cchunk]     = bl0;
            __syncthreads();

            frag_wmma(0);
        }
    }

    // ---- epilogue: sum C^2 over this wave's 32 M values per column --------
    const int psdB = mBase >> 12;          // uniform across block
    const int rB   = (mBase >> 9) & 7;     // uniform across block
#pragma unroll
    for (int sn = 0; sn < 2; ++sn) {
        float s = 0.0f;
#pragma unroll
        for (int sm = 0; sm < 2; ++sm)
#pragma unroll
            for (int e = 0; e < 8; ++e) {
                float c = acc[sm][sn][e];
                s += c * c;
            }
        s += __shfl_xor(s, 16, 32);        // combine lane halves (M 0-7 / 8-15)
        if (lane < 16) {
            const int col = nBase + wn * 32 + sn * 16 + la;
            atomicAdd(&S[psdB * 7680 + rB * 960 + col], s);
        }
    }
}

// --- kernel 4: pair cos/sin, ratio, masked sum, final scalar ---------------
__global__ void reduce_loss_kernel(const float* __restrict__ S,
                                   float* __restrict__ out) {
    __shared__ float red[256];
    float acc = 0.0f;
    for (int f = threadIdx.x; f < 479; f += 256) {
#pragma unroll
        for (int r = 0; r < 8; ++r) {
            float nre = S[r * 960 + 2 * f];
            float nim = S[r * 960 + 2 * f + 1];
            float dre = S[7680 + r * 960 + 2 * f];
            float dIm = S[7680 + r * 960 + 2 * f + 1];
            acc += (nre + nim) / (dre + dIm);
        }
    }
    red[threadIdx.x] = acc;
    __syncthreads();
    for (int off = 128; off > 0; off >>= 1) {
        if ((int)threadIdx.x < off) red[threadIdx.x] += red[threadIdx.x + off];
        __syncthreads();
    }
    // loss_i = sum_f ratio / 480 ; answer = mean(loss[8:16]) * 16 = 2 * sum
    if (threadIdx.x == 0) out[0] = red[0] * (2.0f / 480.0f);
}

extern "C" void kernel_launch(void* const* d_in, const int* in_sizes, int n_in,
                              void* d_out, int out_size, void* d_ws,
                              size_t ws_size, hipStream_t stream) {
    (void)in_sizes; (void)n_in; (void)out_size;
    const float* pred   = (const float*)d_in[0];
    const float* target = (const float*)d_in[1];
    float* out = (float*)d_out;

    char* ws = (char*)d_ws;
    __bf16* Bhi = (__bf16*)(ws + WS_OFF_BHI);
    __bf16* Blo = (__bf16*)(ws + WS_OFF_BLO);
    float*  S   = (float*)(ws + WS_OFF_S);
    float*  win = (float*)(ws + WS_OFF_WIN);
    __bf16* Ahi = (__bf16*)(ws + WS_OFF_AHI);
    __bf16* Alo = (__bf16*)(ws + WS_OFF_ALO);

    init_ws_kernel<<<60, 256, 0, stream>>>(S, win);
    build_basis_kernel<<<(960 * 4096) / 256, 256, 0, stream>>>(Bhi, Blo);

    dim3 grid(15, 64);
    if (ws_size >= WS_NEED_PRE) {
        // Fast path: split frames once (amortized over 15 N-tiles); the
        // 128 MB split-A working set stays resident in the 192 MB L2, and
        // tiles are DMA'd into LDS asynchronously, overlapping the WMMAs.
        prep_frames_kernel<<<8192, 256, 0, stream>>>(pred, target, win, Ahi, Alo);
        dft_gemm_kernel<true><<<grid, 256, 0, stream>>>(
            pred, target, Bhi, Blo, win, Ahi, Alo, S);
    } else {
        dft_gemm_kernel<false><<<grid, 256, 0, stream>>>(
            pred, target, Bhi, Blo, win, Ahi, Alo, S);
    }
    reduce_loss_kernel<<<1, 256, 0, stream>>>(S, out);
}
